// GIN_15925738733670
// MI455X (gfx1250) — matrix operand order repaired and checked by reading, not compiled, not run on hardware
//
#include <hip/hip_runtime.h>

#define NN 100000
#define NE 1600000
#define FEATS 128

typedef _Float16 half8  __attribute__((ext_vector_type(8)));
typedef _Float16 half16 __attribute__((ext_vector_type(16)));
typedef float    f32x8  __attribute__((ext_vector_type(8)));

// ---------------- zero fill ----------------
__global__ void gin_zero_f32(float* __restrict__ p, long n) {
    long i = (long)blockIdx.x * blockDim.x + threadIdx.x;
    long stride = (long)gridDim.x * blockDim.x;
    for (; i < n; i += stride) p[i] = 0.0f;
}

// ---------------- scatter-add aggregation ----------------
// One edge per wave32: float4 gather per lane + 4 hardware f32 atomics.
__global__ __launch_bounds__(256) void gin_scatter_add(
        const float* __restrict__ feat, const int* __restrict__ src,
        const int* __restrict__ dst, float* __restrict__ agg, int nEdges) {
    const int lane = threadIdx.x & 31;
    const int wave = threadIdx.x >> 5;
    const int e = blockIdx.x * (blockDim.x >> 5) + wave;
    if (e >= nEdges) return;
    const int s = src[e];
    const int d = dst[e];
    const float4 v = ((const float4*)(feat + (long)s * FEATS))[lane];
    float* out = agg + (long)d * FEATS + lane * 4;
    unsafeAtomicAdd(out + 0, v.x);
    unsafeAtomicAdd(out + 1, v.y);
    unsafeAtomicAdd(out + 2, v.z);
    unsafeAtomicAdd(out + 3, v.w);
}

// ---------------- fused GIN MLP: out = [ReLU]( ReLU((1+eps)x+agg @ Wa + ba) @ Wb + bb ) ----
// Block: 128 threads (4 waves), tile = 64 rows. Each wave owns a 16-row strip.
template<int K1, int N1, int N2, bool RELU_OUT>
__global__ __launch_bounds__(128) void gin_mlp(
        const float* __restrict__ base, const float* __restrict__ agg,
        const float* __restrict__ epsp,
        const float* __restrict__ Wa, const float* __restrict__ ba,
        const float* __restrict__ Wb, const float* __restrict__ bb,
        float* __restrict__ out, int nRows) {
    constexpr int MT = 64;
    constexpr int WMAX = (K1 * N1 > N1 * N2) ? K1 * N1 : N1 * N2;
    extern __shared__ __align__(16) char smem[];
    _Float16* zs = (_Float16*)smem;        // [MT][K1]  f16 activations
    _Float16* wt = zs + MT * K1;           // [WMAX]    f16 transposed weights [n][k]
    _Float16* ts = wt + WMAX;              // [MT][N1]  f16 hidden

    const int tid  = threadIdx.x;
    const int lane = tid & 31;
    const int wave = tid >> 5;
    const int r0   = blockIdx.x * MT;
    const float epk = 1.0f + *epsp;

    // stage z = (1+eps)*base + agg -> f16 LDS
    for (int i = tid; i < MT * K1 / 4; i += 128) {
        const int m = (i * 4) / K1, k = (i * 4) % K1;
        const int row = r0 + m;
        float4 xv = make_float4(0.f, 0.f, 0.f, 0.f), av = xv;
        if (row < nRows) {
            xv = ((const float4*)base)[(long)row * (K1 / 4) + k / 4];
            av = ((const float4*)agg )[(long)row * (K1 / 4) + k / 4];
        }
        zs[m * K1 + k + 0] = (_Float16)(epk * xv.x + av.x);
        zs[m * K1 + k + 1] = (_Float16)(epk * xv.y + av.y);
        zs[m * K1 + k + 2] = (_Float16)(epk * xv.z + av.z);
        zs[m * K1 + k + 3] = (_Float16)(epk * xv.w + av.w);
    }
    // stage Wa transposed: wt[n*K1+k] = Wa[k*N1+n]
    for (int i = tid; i < K1 * N1; i += 128) {
        const int k = i / N1, n = i % N1;
        wt[n * K1 + k] = (_Float16)Wa[i];
    }
    __syncthreads();

    const int mBase = wave * 16;     // this wave's 16-row strip
    const int cn = lane & 15;        // A: m index / B,C: n index
    const int hs = lane >> 4;        // half-wave select

    // GEMM1: t = ReLU(z @ Wa + ba)
    {
        f32x8 acc[N1 / 16];
        #pragma unroll
        for (int nt = 0; nt < N1 / 16; ++nt)
            #pragma unroll
            for (int r = 0; r < 8; ++r) acc[nt][r] = 0.0f;
        #pragma unroll
        for (int kc = 0; kc < K1; kc += 32) {
            union { half16 v; half8 h[2]; } a;
            const _Float16* ap = &zs[(mBase + cn) * K1 + kc + hs * 16];
            a.h[0] = *(const half8*)ap;           // K = kc + hs*16 .. +7
            a.h[1] = *(const half8*)(ap + 8);     // K = kc + hs*16 + 8 .. +15
            #pragma unroll
            for (int nt = 0; nt < N1 / 16; ++nt) {
                half16 b = *(const half16*)&wt[(nt * 16 + cn) * K1 + kc + hs * 16];
                acc[nt] = __builtin_amdgcn_wmma_f32_16x16x32_f16(
                    false, a.v, false, b, (short)0, acc[nt], false, false);
            }
        }
        #pragma unroll
        for (int nt = 0; nt < N1 / 16; ++nt) {
            const float bias = ba[nt * 16 + cn];
            #pragma unroll
            for (int r = 0; r < 8; ++r) {
                float v = acc[nt][r] + bias;
                v = fmaxf(v, 0.0f);
                ts[(mBase + 8 * hs + r) * N1 + nt * 16 + cn] = (_Float16)v;
            }
        }
    }
    __syncthreads();
    // stage Wb transposed (K=N1, N=N2), reusing the weight buffer
    for (int i = tid; i < N1 * N2; i += 128) {
        const int k = i / N2, n = i % N2;
        wt[n * N1 + k] = (_Float16)Wb[i];
    }
    __syncthreads();

    // GEMM2: out = t @ Wb + bb  (optional ReLU), stream f32 to global
    {
        f32x8 acc[N2 / 16];
        #pragma unroll
        for (int nt = 0; nt < N2 / 16; ++nt)
            #pragma unroll
            for (int r = 0; r < 8; ++r) acc[nt][r] = 0.0f;
        #pragma unroll
        for (int kc = 0; kc < N1; kc += 32) {
            union { half16 v; half8 h[2]; } a;
            const _Float16* ap = &ts[(mBase + cn) * N1 + kc + hs * 16];
            a.h[0] = *(const half8*)ap;
            a.h[1] = *(const half8*)(ap + 8);
            #pragma unroll
            for (int nt = 0; nt < N2 / 16; ++nt) {
                half16 b = *(const half16*)&wt[(nt * 16 + cn) * N1 + kc + hs * 16];
                acc[nt] = __builtin_amdgcn_wmma_f32_16x16x32_f16(
                    false, a.v, false, b, (short)0, acc[nt], false, false);
            }
        }
        #pragma unroll
        for (int nt = 0; nt < N2 / 16; ++nt) {
            const float bias = bb[nt * 16 + cn];
            #pragma unroll
            for (int r = 0; r < 8; ++r) {
                const int row = r0 + mBase + 8 * hs + r;
                if (row < nRows) {
                    float v = acc[nt][r] + bias;
                    if (RELU_OUT) v = fmaxf(v, 0.0f);
                    out[(long)row * N2 + nt * 16 + cn] = v;
                }
            }
        }
    }
}

extern "C" void kernel_launch(void* const* d_in, const int* in_sizes, int n_in,
                              void* d_out, int out_size, void* d_ws, size_t ws_size,
                              hipStream_t stream) {
    const float* x    = (const float*)d_in[0];
    const int*   ei   = (const int*)  d_in[1];
    const float* eps1 = (const float*)d_in[2];
    const float* W1a  = (const float*)d_in[3];
    const float* b1a  = (const float*)d_in[4];
    const float* W1b  = (const float*)d_in[5];
    const float* b1b  = (const float*)d_in[6];
    const float* eps2 = (const float*)d_in[7];
    const float* W2a  = (const float*)d_in[8];
    const float* b2a  = (const float*)d_in[9];
    const float* W2b  = (const float*)d_in[10];
    const float* b2b  = (const float*)d_in[11];
    const int* src = ei;
    const int* dst = ei + NE;

    float* agg = (float*)d_ws;                       // [NN,128]
    float* h1  = agg + (size_t)NN * FEATS;           // [NN,128]
    float* outp = (float*)d_out;                     // [NN,64]

    const long aggN = (long)NN * FEATS;
    const int  edgeBlocks = (NE + 7) / 8;            // 8 edges (waves) per block
    const int  mlpBlocks  = (NN + 63) / 64;

    // Layer 1
    gin_zero_f32<<<2048, 256, 0, stream>>>(agg, aggN);
    gin_scatter_add<<<edgeBlocks, 256, 0, stream>>>(x, src, dst, agg, NE);
    gin_mlp<128, 128, 128, true><<<mlpBlocks, 128, 65536, stream>>>(
        x, agg, eps1, W1a, b1a, W1b, b1b, h1, NN);

    // Layer 2
    gin_zero_f32<<<2048, 256, 0, stream>>>(agg, aggN);
    gin_scatter_add<<<edgeBlocks, 256, 0, stream>>>(h1, src, dst, agg, NE);
    gin_mlp<128, 64, 64, false><<<mlpBlocks, 128, 40960, stream>>>(
        h1, agg, eps2, W2a, b2a, W2b, b2b, outp, NN);
}
